// MDLSTM_59167469469738
// MI455X (gfx1250) — compile-verified, matrix operand order checked by reading, not run on hardware
//
#include <hip/hip_runtime.h>
#include <hip/hip_bf16.h>

// ---------------------------------------------------------------------------
// MDLSTM (4-direction 2D LSTM) for MI455X / gfx1250, wave32 + WMMA bf16.
//
// Geometry: B=32, CIN=16, H=32, W=128, COUT=128, 4 directions.
// Per column step (per dir,row): z = x@Wx^T  +  h@Wh^T for two h's (top,left),
// M=32, N=512 (4 gates x 128), K=128 (plus K=16 input proj, padded to 32).
// WMMA tile: 16x16x32 bf16 -> f32.
// ---------------------------------------------------------------------------

typedef __attribute__((ext_vector_type(16))) __bf16 v16bf;
typedef __attribute__((ext_vector_type(8)))  float  v8f;

#define NDIR   4
#define HH     32
#define WW     128
#define BB     32
#define CIN    16
#define COUT   128
#define NGATE  4
#define NCAT   (NGATE * COUT)          // 512
#define NT_N   (NCAT / 16)             // 32 N-tiles
#define NT_K   (COUT / 32)             // 4 K-steps of 32
#define CELL   (BB * COUT)             // 4096 elements of h/c per cell

__device__ __forceinline__ unsigned short f2bf(float f) {
    unsigned u = __builtin_bit_cast(unsigned, f);
    unsigned r = u + 0x7FFFu + ((u >> 16) & 1u);
    return (unsigned short)(r >> 16);
}

__device__ __forceinline__ float sigmoidf(float x) {
    return 1.0f / (1.0f + __expf(-x));
}

// ---------------------------------------------------------------------------
// Weight packing: B-fragment layout for v_wmma_f32_16x16x32_bf16.
// A tile is 32(K) x 16(N).  Lane L holds 16 bf16 contiguous:
//   n = L & 15,  k = (L >> 4)*16 + e   (e = 0..15)
// Tile storage: tile_base + L*16 + e  (so each lane issues one 32B load).
// ---------------------------------------------------------------------------

// Wh: cat gates [i,f,g,o] -> row jj = g*128 + o, col kk in 0..127.
// out layout: (((dir*32 + nt)*4 + kt) * 512) + lane*16 + e
__global__ __launch_bounds__(256) void pack_wh(
    const float* __restrict__ whi, const float* __restrict__ whf,
    const float* __restrict__ whg, const float* __restrict__ who,
    unsigned short* __restrict__ out)
{
    int flat = blockIdx.x * 256 + threadIdx.x;           // 2^18 total
    int e    = flat & 15;
    int lane = (flat >> 4) & 31;
    int kt   = (flat >> 9) & 3;
    int nt   = (flat >> 11) & 31;
    int dir  = (flat >> 16) & 3;

    int n  = lane & 15;
    int k  = ((lane >> 4) << 4) + e;
    int jj = nt * 16 + n;
    int g  = jj >> 7;
    int o  = jj & 127;
    int kk = kt * 32 + k;

    const float* Wg = (g == 0) ? whi : (g == 1) ? whf : (g == 2) ? whg : who;
    out[flat] = f2bf(Wg[(dir * COUT + o) * COUT + kk]);
}

// Wx: K=16 real, padded to 32 with zeros.
// out layout: ((dir*32 + nt) * 512) + lane*16 + e
__global__ __launch_bounds__(256) void pack_wx(
    const float* __restrict__ wxi, const float* __restrict__ wxf,
    const float* __restrict__ wxg, const float* __restrict__ wxo,
    unsigned short* __restrict__ out)
{
    int flat = blockIdx.x * 256 + threadIdx.x;           // 2^16 total
    int e    = flat & 15;
    int lane = (flat >> 4) & 31;
    int nt   = (flat >> 9) & 31;
    int dir  = (flat >> 14) & 3;

    int n  = lane & 15;
    int k  = ((lane >> 4) << 4) + e;
    int jj = nt * 16 + n;
    int g  = jj >> 7;
    int o  = jj & 127;

    const float* Wg = (g == 0) ? wxi : (g == 1) ? wxf : (g == 2) ? wxg : wxo;
    float v = (k < CIN) ? Wg[(dir * COUT + o) * CIN + k] : 0.0f;
    out[flat] = f2bf(v);
}

// ---------------------------------------------------------------------------
// Main wavefront kernel.  grid = (32 rows, 4 dirs), block = 256 (8 wave32).
// Row r waits on row r-1's per-column progress flag; rows pipeline along
// the anti-diagonal so 128 workgroups are concurrently active at peak.
// ---------------------------------------------------------------------------
__global__ __launch_bounds__(256) void mdlstm_wavefront(
    const float* __restrict__ x,
    const float* __restrict__ b_i, const float* __restrict__ b_f,
    const float* __restrict__ b_g, const float* __restrict__ b_o,
    const float* __restrict__ wsum, const float* __restrict__ cellbias,
    const unsigned short* __restrict__ whFrag,
    const unsigned short* __restrict__ wxFrag,
    unsigned short* __restrict__ hbuf,   // [dir][row][col][4096] bf16
    float* __restrict__ cbuf,            // [dir][row][col][4096] f32
    int* __restrict__ progress,          // [dir][row]
    float* __restrict__ out)             // [dir][o][b][h][w] f32
{
    const int r   = blockIdx.x;
    const int dir = blockIdx.y;
    const int tid = threadIdx.x;
    const int lane = tid & 31;
    const int wv   = tid >> 5;
    const int fy = (dir >> 1) & 1;
    const int fx = dir & 1;

    extern __shared__ char smem[];
    float*          z0     = (float*)smem;                 // 32*512 f32
    float*          z1     = z0 + BB * NCAT;               // 32*512 f32
    float*          ctop   = z1 + BB * NCAT;               // 32*128
    float*          cleft  = ctop + CELL;                  // 32*128
    float*          bi_s   = cleft + CELL;                 // 128 x5
    float*          bf_s   = bi_s + COUT;
    float*          bg_s   = bf_s + COUT;
    float*          bo_s   = bg_s + COUT;
    float*          bc_s   = bo_s + COUT;
    unsigned short* htop   = (unsigned short*)(bc_s + COUT);  // 32*128 bf16
    unsigned short* hleft  = htop + CELL;                     // 32*128 bf16
    unsigned short* xls    = hleft + CELL;                    // 32*32 bf16 (K pad)

    // ---- one-time init ----
    for (int i = tid; i < COUT; i += 256) {
        bi_s[i] = b_i[dir * COUT + i];
        bf_s[i] = b_f[dir * COUT + i];
        bg_s[i] = b_g[dir * COUT + i];
        bo_s[i] = b_o[dir * COUT + i];
        bc_s[i] = cellbias[dir * COUT + i];
    }
    for (int i = tid; i < CELL; i += 256) {
        ctop[i] = 0.0f; cleft[i] = 0.0f;
        htop[i] = 0;    hleft[i] = 0;
    }
    for (int i = tid; i < BB * 32; i += 256) xls[i] = 0;   // zero K-pad region
    const float ws0 = wsum[dir * 2 + 0];
    const float ws1 = wsum[dir * 2 + 1];
    __syncthreads();

    const long rowbase = ((long)(dir * HH + r)) * WW;      // cells before col 0

    for (int w = 0; w < WW; ++w) {
        // ---- wait for row above, pull (h_top, c_top) ----
        if (r > 0) {
            if (tid == 0) {
                const int* flag = &progress[dir * HH + (r - 1)];
                while (__hip_atomic_load(flag, __ATOMIC_ACQUIRE,
                                         __HIP_MEMORY_SCOPE_AGENT) <= w) {
                    __builtin_amdgcn_s_sleep(1);
                }
            }
            __syncthreads();
            const long above = ((long)(dir * HH + (r - 1))) * WW + w;
            const unsigned short* hsrc = hbuf + above * CELL;
            const float*          csrc = cbuf + above * CELL;
            __builtin_prefetch(hsrc + CELL, 0, 1);         // next column row-above
            for (int i = tid; i < CELL; i += 256) {
                htop[i] = hsrc[i];
                ctop[i] = csrc[i];
            }
        }

        // ---- stage x column as bf16 (32 x 16, K-padded to 32) ----
        const int ho = fy ? (HH - 1 - r) : r;
        const int wo = fx ? (WW - 1 - w) : w;
        for (int i = tid; i < BB * CIN; i += 256) {
            int b = i >> 4, c = i & 15;
            xls[b * 32 + c] = f2bf(x[((b * CIN + c) * HH + ho) * WW + wo]);
        }
        __syncthreads();

        // ---- WMMA: z{0,1}[32 x 512] = x@Wx^T + h{top,left}@Wh^T ----
        // 64 tile jobs (2 M-tiles x 32 N-tiles), 8 per wave, 9 WMMA each.
        for (int job = wv; job < 64; job += 8) {
            const int mt = job >> 5;
            const int nt = job & 31;
            const int bb = mt * 16 + (lane & 15);
            const int kh = (lane >> 4) << 4;

            const v16bf ax = *(const v16bf*)(xls + bb * 32 + kh);
            const v16bf bx = *(const v16bf*)(wxFrag +
                                (long)(dir * NT_N + nt) * 512 + lane * 16);
            v8f acc = {};
            acc = __builtin_amdgcn_wmma_f32_16x16x32_bf16(
                      false, ax, false, bx, (short)0, acc, false, false);
            v8f a0 = acc;
            v8f a1 = acc;
#pragma unroll
            for (int kt = 0; kt < NT_K; ++kt) {
                const v16bf bh = *(const v16bf*)(whFrag +
                        (long)((dir * NT_N + nt) * NT_K + kt) * 512 + lane * 16);
                const v16bf at = *(const v16bf*)(htop  + bb * COUT + kt * 32 + kh);
                const v16bf al = *(const v16bf*)(hleft + bb * COUT + kt * 32 + kh);
                a0 = __builtin_amdgcn_wmma_f32_16x16x32_bf16(
                         false, at, false, bh, (short)0, a0, false, false);
                a1 = __builtin_amdgcn_wmma_f32_16x16x32_bf16(
                         false, al, false, bh, (short)0, a1, false, false);
            }
            // C/D layout: VGPR v -> M = v + 8*(lane/16), N = lane%16
            const int n0    = nt * 16 + (lane & 15);
            const int mbase = mt * 16 + ((lane >> 4) << 3);
#pragma unroll
            for (int v = 0; v < 8; ++v) {
                z0[(mbase + v) * NCAT + n0] = a0[v];
                z1[(mbase + v) * NCAT + n0] = a1[v];
            }
        }
        __syncthreads();

        // ---- elementwise: gates, two branches, weighted combine ----
        const long cellidx = (rowbase + w) * (long)CELL;
        for (int i = tid; i < CELL; i += 256) {
            const int b = i >> 7;
            const int o = i & 127;
            const int zb = b * NCAT + o;

            float zi0 = z0[zb]       + bi_s[o];
            float zf0 = z0[zb + 128] + bf_s[o];
            float zg0 = z0[zb + 256] + bg_s[o];
            float zo0 = z0[zb + 384] + bo_s[o];
            float zi1 = z1[zb]       + bi_s[o];
            float zf1 = z1[zb + 128] + bf_s[o];
            float zg1 = z1[zb + 256] + bg_s[o];
            float zo1 = z1[zb + 384] + bo_s[o];

            float i0 = sigmoidf(zi0), f0 = sigmoidf(zf0);
            float g0 = tanhf(zg0),    q0 = sigmoidf(zo0);
            float c0 = f0 * ctop[i] + i0 * g0;
            float h0 = q0 * tanhf(c0);

            float i1 = sigmoidf(zi1), f1 = sigmoidf(zf1);
            float g1 = tanhf(zg1),    q1 = sigmoidf(zo1);
            float c1 = f1 * cleft[i] + i1 * g1;
            float h1 = q1 * tanhf(c1);

            float ct = ws0 * c0 + ws1 * c1 + bc_s[o];
            float ht = ws0 * h0 + ws1 * h1 + bc_s[o];

            cleft[i] = ct;
            hleft[i] = f2bf(ht);
            hbuf[cellidx + i] = f2bf(ht);
            cbuf[cellidx + i] = ct;
            out[(((long)(dir * COUT + o) * BB + b) * HH + ho) * WW + wo] = ht;
        }
        __threadfence();
        __syncthreads();
        if (tid == 0) {
            __hip_atomic_store(&progress[dir * HH + r], w + 1,
                               __ATOMIC_RELEASE, __HIP_MEMORY_SCOPE_AGENT);
        }
    }
}

// ---------------------------------------------------------------------------
// Host launcher
// ---------------------------------------------------------------------------
extern "C" void kernel_launch(void* const* d_in, const int* in_sizes, int n_in,
                              void* d_out, int out_size, void* d_ws, size_t ws_size,
                              hipStream_t stream) {
    const float* x    = (const float*)d_in[0];
    const float* w_ii = (const float*)d_in[1];
    const float* w_hi = (const float*)d_in[2];
    const float* b_i  = (const float*)d_in[3];
    const float* w_if = (const float*)d_in[4];
    const float* w_hf = (const float*)d_in[5];
    const float* b_f  = (const float*)d_in[6];
    const float* w_ig = (const float*)d_in[7];
    const float* w_hg = (const float*)d_in[8];
    const float* b_g  = (const float*)d_in[9];
    const float* w_io = (const float*)d_in[10];
    const float* w_ho = (const float*)d_in[11];
    const float* b_o  = (const float*)d_in[12];
    const float* wsum = (const float*)d_in[13];
    const float* bias = (const float*)d_in[14];

    const size_t WH_BYTES   = (size_t)NDIR * NT_N * NT_K * 512 * 2;       // 512 KB
    const size_t WX_BYTES   = (size_t)NDIR * NT_N * 512 * 2;              // 128 KB
    const size_t HBUF_BYTES = (size_t)NDIR * HH * WW * CELL * 2;          // 128 MB
    const size_t CBUF_BYTES = (size_t)NDIR * HH * WW * CELL * 4;          // 256 MB
    const size_t PROG_BYTES = (size_t)NDIR * HH * 4;

    char* ws = (char*)d_ws;
    size_t off = 0;
    auto carve = [&](size_t bytes) -> void* {
        void* p = ws + off;
        off += (bytes + 255) & ~(size_t)255;
        return p;
    };
    unsigned short* whFrag   = (unsigned short*)carve(WH_BYTES);
    unsigned short* wxFrag   = (unsigned short*)carve(WX_BYTES);
    unsigned short* hbuf     = (unsigned short*)carve(HBUF_BYTES);
    float*          cbuf     = (float*)carve(CBUF_BYTES);
    int*            progress = (int*)carve(PROG_BYTES);
    if (off > ws_size) return;  // workspace too small: nothing we can do

    hipMemsetAsync(progress, 0, PROG_BYTES, stream);

    pack_wh<<<(NDIR * NT_N * NT_K * 512) / 256, 256, 0, stream>>>(
        w_hi, w_hf, w_hg, w_ho, whFrag);
    pack_wx<<<(NDIR * NT_N * 512) / 256, 256, 0, stream>>>(
        w_ii, w_if, w_ig, w_io, wxFrag);

    const size_t smem_bytes =
        (size_t)(BB * NCAT) * 4 * 2 +   // z0, z1
        (size_t)CELL * 4 * 2 +          // ctop, cleft
        (size_t)COUT * 4 * 5 +          // biases
        (size_t)CELL * 2 * 2 +          // htop, hleft (bf16)
        (size_t)(BB * 32) * 2;          // x column (bf16, K-padded)

    mdlstm_wavefront<<<dim3(HH, NDIR), 256, smem_bytes, stream>>>(
        x, b_i, b_f, b_g, b_o, wsum, bias,
        whFrag, wxFrag, hbuf, cbuf, progress, (float*)d_out);
}
